// TriangleMultiplicationOutgoing_31645319037641
// MI455X (gfx1250) — matrix operand order repaired and checked by reading, not compiled, not run on hardware
//
#include <hip/hip_runtime.h>
#include <hip/hip_bf16.h>

// ---------------------------------------------------------------------------
// TriangleMultiplicationOutgoing for MI455X (gfx1250), bf16 WMMA pipeline.
//   B=1, N=512, D=C=128.  ~86 GFLOP vs ~600MB min HBM traffic: balanced at
//   23.3 TB/s, so run all GEMMs as bf16 WMMA, fuse LN1+sigmoid+mask into the
//   projection epilogues, keep left/right (64MB each bf16, [c][row][k]
//   row-major) L2-resident for the triangle einsum, and fuse LN2+W_op+gated
//   residual into the final pass. All WMMA fragments are fed by b128 loads
//   (2x ds/global_load_b128 per fragment), LDS rows padded +8 bf16 so the 16
//   fragment rows land on distinct bank groups. Sigmoid uses v_rcp_f32
//   (bf16-bound output), avoiding IEEE div chains in the epilogues.
// ---------------------------------------------------------------------------

typedef __bf16 bf16;
typedef __attribute__((ext_vector_type(16))) __bf16 v16bf;
typedef __attribute__((ext_vector_type(8)))  float  v8f;

#define NN 512
#define DD 128
#define PLANE (512 * 512)

// Fast sigmoid: v_exp_f32 + v_rcp_f32 (result is stored as bf16; ~1ulp rcp
// is far below bf16 quantization error). Avoids v_div_scale/fmas/fixup chains.
__device__ __forceinline__ float sigmoidf_(float x) {
    return __builtin_amdgcn_rcpf(1.0f + __expf(-x));
}

// D = A*B + C (emits v_wmma_f32_16x16x32_bf16)
__device__ __forceinline__ v8f wmma_bf16(v16bf a, v16bf b, v8f c) {
    return __builtin_amdgcn_wmma_f32_16x16x32_bf16(
        false, a, false, b, (short)0, c, false, false);
}

// A fragment: 16x32 (MxK) bf16, row-major, row stride ldk (multiple of 8).
// Lane l: M = l%16; VGPR 0-3 hold K = 8*(l/16)+0..7, VGPR 4-7 hold +16.
// Exactly two 16B loads per lane.
__device__ __forceinline__ v16bf load_a_frag(const bf16* __restrict__ base,
                                             int ldk, int lane) {
    const bf16* rp = base + (lane & 15) * ldk + (((lane >> 4) & 1) << 3);
    union { v16bf v; uint4 q[2]; } u;
    u.q[0] = *(const uint4*)(rp);
    u.q[1] = *(const uint4*)(rp + 16);
    return u.v;
}

// B fragment: 32x16 (KxN) bf16 where the tile is stored N-major (row n holds
// the K-run), row stride ldk. Lane l: N = l%16; VGPR 0-3 hold K = 8*(l/16)+0..7,
// VGPR 4-7 hold +16 (same order as A). Exactly two 16B loads per lane.
__device__ __forceinline__ v16bf load_b_frag(const bf16* __restrict__ base,
                                             int ldk, int lane) {
    const bf16* rp = base + (lane & 15) * ldk + (((lane >> 4) & 1) << 3);
    union { v16bf v; uint4 q[2]; } u;
    u.q[0] = *(const uint4*)(rp);
    u.q[1] = *(const uint4*)(rp + 16);
    return u.v;
}

// ---------------------------------------------------------------------------
// Kernel 0: pack all weights into one column-major bf16 table Wt[col][k],
// k contiguous, so every B fragment in kernels 1/3 is 2x global_load_b128
// from a ~200KB L0/L2-hot buffer (no f32 loads / cvt in hot loops).
//   cols 0..255: W_lr | 256..511: W_gate | 512..639: W_og | 640..767: W_op
// ---------------------------------------------------------------------------
__global__ __launch_bounds__(128) void pack_weights_kernel(
    const float* __restrict__ W_lr, const float* __restrict__ W_gate,
    const float* __restrict__ W_og, const float* __restrict__ W_op,
    bf16* __restrict__ Wt) {
    const int col = blockIdx.x;      // 0..767
    const int k   = threadIdx.x;     // 0..127
    float v;
    if (col < 256)      v = W_lr[k * 256 + col];
    else if (col < 512) v = W_gate[k * 256 + (col - 256)];
    else if (col < 640) v = W_og[k * 128 + (col - 512)];
    else                v = W_op[k * 128 + (col - 640)];
    Wt[(long)col * DD + k] = (bf16)v;
}

// ---------------------------------------------------------------------------
// Kernel 1: LN1 + (Z@W_lr + b)*mask*sig(Z@W_gate + b) and sig(Z@W_og + b),
// 16 rows/block, 4 waves x 6 wave-uniform 16x16 tile tasks.
// Outputs: Lt[c][i][k], Rt[c][j][k] (bf16, einsum-ready), G[r][d] (bf16).
// ---------------------------------------------------------------------------
#define ZLD 136   // 128 + 8 pad: 16B-aligned rows, bank-conflict-free frags
__global__ __launch_bounds__(128) void prep_kernel(
    const float* __restrict__ Zraw, const float* __restrict__ Zmask,
    const float* __restrict__ ln1_g, const float* __restrict__ ln1_b,
    const bf16* __restrict__ Wt,
    const float* __restrict__ b_lr, const float* __restrict__ b_gate,
    const float* __restrict__ b_og,
    bf16* __restrict__ Lt, bf16* __restrict__ Rt, bf16* __restrict__ G) {
    __shared__ float zf[16][DD];
    __shared__ bf16  za[16 * ZLD];
    __shared__ float mu_s[16], rs_s[16], mk_s[16];

    const int  t  = threadIdx.x;
    const long r0 = (long)blockIdx.x * 16;   // 16 rows of flattened (i,k)
    const int  ii = (int)(r0 >> 9);          // i (left) == j (right) index
    const int  k0 = (int)(r0 & 511);         // k base

    if (blockIdx.x + 1 < gridDim.x)
        __builtin_prefetch(&Zraw[(r0 + 16) * DD + t], 0, 1);
#pragma unroll
    for (int e = 0; e < 16; ++e) zf[e][t] = Zraw[(r0 + e) * DD + t];
    __syncthreads();

    if (t < 16) {
        float s1 = 0.f, s2 = 0.f;
#pragma unroll 4
        for (int c = 0; c < DD; ++c) {
            float v = zf[t][c];
            s1 += v; s2 += v * v;
        }
        float mu  = s1 * (1.0f / DD);
        float var = s2 * (1.0f / DD) - mu * mu;
        mu_s[t] = mu;
        rs_s[t] = rsqrtf(var + 1e-5f);
        mk_s[t] = Zmask[r0 + t];
    }
    __syncthreads();
#pragma unroll
    for (int e = 0; e < 16; ++e) {
        float v = (zf[e][t] - mu_s[e]) * rs_s[e] * ln1_g[t] + ln1_b[t];
        za[e * ZLD + t] = (bf16)v;
    }
    __syncthreads();

    const int w = t >> 5, lane = t & 31, n = lane & 15;
    for (int s = 0; s < 6; ++s) {
        const int tid = w * 6 + s;           // 24 tasks: 16 lr|gate, 8 og
        if (tid < 16) {
            const int c0 = tid * 16;
            v8f acc_l = {}; v8f acc_g = {};
#pragma unroll
            for (int kc = 0; kc < DD; kc += 32) {
                v16bf a = load_a_frag(za + kc, ZLD, lane);
                acc_l = wmma_bf16(a, load_b_frag(Wt + (long)c0 * DD + kc, DD, lane), acc_l);
                acc_g = wmma_bf16(a, load_b_frag(Wt + (long)(256 + c0) * DD + kc, DD, lane), acc_g);
            }
            const float bl = b_lr[c0 + n], bg = b_gate[c0 + n];
            bf16* dst = (c0 < 128)
                ? (Lt + (long)(c0 + n) * PLANE + (long)ii * NN)
                : (Rt + (long)(c0 - 128 + n) * PLANE + (long)ii * NN);
#pragma unroll
            for (int v = 0; v < 8; ++v) {
                const int   M   = v + ((lane >> 4) << 3);
                const float val = (acc_l[v] + bl) * mk_s[M] * sigmoidf_(acc_g[v] + bg);
                dst[k0 + M] = (bf16)val;
            }
        } else {
            const int c0 = (tid - 16) * 16;
            v8f acc = {};
#pragma unroll
            for (int kc = 0; kc < DD; kc += 32)
                acc = wmma_bf16(load_a_frag(za + kc, ZLD, lane),
                                load_b_frag(Wt + (long)(512 + c0) * DD + kc, DD, lane), acc);
            const float bo = b_og[c0 + n];
#pragma unroll
            for (int v = 0; v < 8; ++v) {
                const int M = v + ((lane >> 4) << 3);
                G[(r0 + M) * DD + c0 + n] = (bf16)sigmoidf_(acc[v] + bo);
            }
        }
    }
}

// ---------------------------------------------------------------------------
// Kernel 2: triangle einsum. Per (c, 64x64 ij tile): P_c = L_c @ R_c^T over
// K=512, staged in four 128-wide LDS phases. Each wave owns a 32x32 quadrant
// (4 accumulators, 2x2 fragment reuse): 64 WMMAs/wave, 4 WMMAs per
// 8 ds_load_b128. L/R (128MB bf16 total) stay resident in the 192MB L2.
// ---------------------------------------------------------------------------
#define KLD 136   // 128 + 8 pad
__global__ __launch_bounds__(128) void triangle_kernel(
    const bf16* __restrict__ Lt, const bf16* __restrict__ Rt,
    float* __restrict__ P) {
    __shared__ bf16 Ls[64 * KLD];   // 17 KB: 64 i-rows x 128 k (+pad)
    __shared__ bf16 Rs[64 * KLD];   // 17 KB: 64 j-rows x 128 k (+pad)

    const int bx = blockIdx.x;               // 128 c * 8 * 8
    const int c  = bx >> 6;
    const int i0 = ((bx >> 3) & 7) * 64;
    const int j0 = (bx & 7) * 64;
    const int t  = threadIdx.x;
    const int w = t >> 5, lane = t & 31;
    const int mi = (w >> 1) * 32, nj = (w & 1) * 32;
    const long cOff = (long)c * PLANE;

    const bf16* lbase = Lt + cOff + (long)i0 * NN;
    const bf16* rbase = Rt + cOff + (long)j0 * NN;

    v8f a00 = {}, a01 = {}, a10 = {}, a11 = {};
    for (int kp = 0; kp < 4; ++kp) {
        const int kbase = kp * 128;
        const uint4* gl = (const uint4*)(lbase + kbase);
        const uint4* gr = (const uint4*)(rbase + kbase);
        // 64 rows x 16 uint4 per buffer, fully coalesced b128 copies.
#pragma unroll
        for (int s = 0; s < 8; ++s) {
            int idx = t + 128 * s;
            int row = idx >> 4, col = idx & 15;
            ((uint4*)(Ls + row * KLD))[col] = gl[row * 64 + col];
            ((uint4*)(Rs + row * KLD))[col] = gr[row * 64 + col];
        }
        if (kp < 3) {  // prefetch next K-phase (global_prefetch_b8)
            __builtin_prefetch(lbase + kbase + 128 + (long)(t & 63) * NN, 0, 1);
            __builtin_prefetch(rbase + kbase + 128 + (long)(t & 63) * NN, 0, 1);
        }
        __syncthreads();
#pragma unroll
        for (int kk = 0; kk < 128; kk += 32) {
            v16bf af0 = load_a_frag(Ls + (mi +  0) * KLD + kk, KLD, lane);
            v16bf af1 = load_a_frag(Ls + (mi + 16) * KLD + kk, KLD, lane);
            v16bf bf0 = load_b_frag(Rs + (nj +  0) * KLD + kk, KLD, lane);
            v16bf bf1 = load_b_frag(Rs + (nj + 16) * KLD + kk, KLD, lane);
            a00 = wmma_bf16(af0, bf0, a00);
            a01 = wmma_bf16(af0, bf1, a01);
            a10 = wmma_bf16(af1, bf0, a10);
            a11 = wmma_bf16(af1, bf1, a11);
        }
        __syncthreads();
    }

    const int n = lane & 15;
    const int Mh = (lane >> 4) << 3;
    float* pout = P + cOff + (long)(i0 + mi) * NN + j0 + nj;
#pragma unroll
    for (int v = 0; v < 8; ++v) {
        const int M = v + Mh;
        pout[(long)M * NN + n]              = a00[v];
        pout[(long)M * NN + 16 + n]         = a01[v];
        pout[(long)(M + 16) * NN + n]       = a10[v];
        pout[(long)(M + 16) * NN + 16 + n]  = a11[v];
    }
}

// ---------------------------------------------------------------------------
// Kernel 3: out = Z_raw + g * (LN2(p) @ W_op + out_bias). 64 positions/block.
// p read transposed (c-plane stride, lane-coalesced within planes); W_op
// fragments come pre-packed from Wt (global b128, L0-hot).
// ---------------------------------------------------------------------------
__global__ __launch_bounds__(128) void out_kernel(
    const float* __restrict__ Zraw, const bf16* __restrict__ G,
    const float* __restrict__ P, const bf16* __restrict__ Wt,
    const float* __restrict__ ln2_g, const float* __restrict__ ln2_b,
    const float* __restrict__ out_bias, float* __restrict__ out) {
    __shared__ bf16  As[64 * ZLD];   // normalized rows, 17 KB
    __shared__ float mu_s[64], rs_s[64];

    const int  t    = threadIdx.x;
    const long pos0 = (long)blockIdx.x * 64;
    const bf16* WtOp = Wt + (long)640 * DD;

    // LN2 stats per position (lanes coalesce within each c-plane).
    if (t < 64) {
        const long pos = pos0 + t;
        float s1 = 0.f, s2 = 0.f;
#pragma unroll 4
        for (int c = 0; c < 128; ++c) {
            float v = P[(long)c * PLANE + pos];
            s1 += v; s2 += v * v;
        }
        float mu  = s1 * (1.0f / 128.f);
        float var = s2 * (1.0f / 128.f) - mu * mu;
        mu_s[t] = mu;
        rs_s[t] = rsqrtf(var + 1e-5f);
    }
    __syncthreads();
    // Normalize -> bf16 A tiles (global reads coalesced per c-plane).
#pragma unroll 4
    for (int s = 0; s < 64; ++s) {
        int idx = t + 128 * s;       // 8192 = 128c x 64 rows
        int c = idx >> 6, row = idx & 63;
        float v = (P[(long)c * PLANE + pos0 + row] - mu_s[row]) * rs_s[row] *
                      ln2_g[c] + ln2_b[c];
        As[row * ZLD + c] = (bf16)v;
    }
    __syncthreads();

    const int w = t >> 5, lane = t & 31, n = lane & 15;
    const int mw = w * 16;
    for (int c0 = 0; c0 < 128; c0 += 16) {
        v8f acc = {};
#pragma unroll
        for (int kc = 0; kc < 128; kc += 32)
            acc = wmma_bf16(load_a_frag(As + mw * ZLD + kc, ZLD, lane),
                            load_b_frag(WtOp + (long)c0 * DD + kc, DD, lane), acc);
        const float ob = out_bias[c0 + n];
#pragma unroll
        for (int v = 0; v < 8; ++v) {
            const int  M    = v + ((lane >> 4) << 3);
            const long oidx = (pos0 + mw + M) * DD + c0 + n;
            out[oidx] = Zraw[oidx] + (float)G[oidx] * (acc[v] + ob);
        }
    }
}

// ---------------------------------------------------------------------------
extern "C" void kernel_launch(void* const* d_in, const int* in_sizes, int n_in,
                              void* d_out, int out_size, void* d_ws,
                              size_t ws_size, hipStream_t stream) {
    const float* Zraw   = (const float*)d_in[0];
    const float* Zmask  = (const float*)d_in[1];
    const float* ln1_g  = (const float*)d_in[2];
    const float* ln1_b  = (const float*)d_in[3];
    const float* W_lr   = (const float*)d_in[4];
    const float* b_lr   = (const float*)d_in[5];
    const float* W_gate = (const float*)d_in[6];
    const float* b_gate = (const float*)d_in[7];
    const float* W_og   = (const float*)d_in[8];
    const float* b_og   = (const float*)d_in[9];
    const float* ln2_g  = (const float*)d_in[10];
    const float* ln2_b  = (const float*)d_in[11];
    const float* W_op   = (const float*)d_in[12];
    const float* out_b  = (const float*)d_in[13];

    // Workspace: Lt 64MB | Rt 64MB | G 64MB | P 128MB | Wt 192KB
    char*  ws = (char*)d_ws;
    bf16*  Lt = (bf16*)(ws);
    bf16*  Rt = (bf16*)(ws + ((size_t)64 << 20));
    bf16*  G  = (bf16*)(ws + ((size_t)128 << 20));
    float* P  = (float*)(ws + ((size_t)192 << 20));
    bf16*  Wt = (bf16*)(ws + ((size_t)320 << 20));

    pack_weights_kernel<<<768, 128, 0, stream>>>(W_lr, W_gate, W_og, W_op, Wt);
    prep_kernel<<<(512 * 512) / 16, 128, 0, stream>>>(
        Zraw, Zmask, ln1_g, ln1_b, Wt, b_lr, b_gate, b_og, Lt, Rt, G);
    triangle_kernel<<<128 * 8 * 8, 128, 0, stream>>>(Lt, Rt, P);
    out_kernel<<<(512 * 512) / 64, 128, 0, stream>>>(
        Zraw, G, P, Wt, ln2_g, ln2_b, out_b, (float*)d_out);
}
